// SeedGNN_hun_85031762526895
// MI455X (gfx1250) — matrix operand order repaired and checked by reading, not compiled, not run on hardware
//
#include <hip/hip_runtime.h>
#include <hip/hip_bf16.h>

#define N 1536
#define NN ((size_t)N * N)
#define HID 8
#define NSEEDS 150
#define TILES (N / 16)            // 96 tiles per dim
#define TILE2 (TILES * TILES)     // 9216 16x16 tiles
#define WPB 8                     // waves per block (256 threads, wave32)

typedef __attribute__((ext_vector_type(16))) _Float16 v16h;
typedef __attribute__((ext_vector_type(8)))  _Float16 v8h;
typedef __attribute__((ext_vector_type(8)))  float    v8f;

// ---------------------------------------------------------------- WMMA helpers

__device__ __forceinline__ v8f wmma_f32_16x16x32(v16h a, v16h b, v8f c) {
  // D = A(16x32 f16) x B(32x16 f16) + C(16x16 f32)
  return __builtin_amdgcn_wmma_f32_16x16x32_f16(
      /*neg_a=*/false, a, /*neg_b=*/false, b,
      /*c_mod=*/(short)0, c, /*reuse_a=*/false, /*reuse_b=*/false);
}

// A-fragment: row-major A, lane holds row m = row0+(lane&15).
// halfs 0..15 -> K = k0 + (lane>>4)*16 + i       (32B contiguous)
__device__ __forceinline__ v16h load_a_frag(const _Float16* __restrict__ A,
                                            int row0, int k0, int lane) {
  const int m  = row0 + (lane & 15);
  const int kb = k0 + ((lane >> 4) << 4);
  return *(const v16h*)(A + (size_t)m * N + kb);
}

// B-fragment from a matrix Bt stored so B[k][n] = Bt[n*N + k]:
// lane holds col n = col0+(lane&15); halfs 0..15 -> K = k0 + (lane>>4)*16 + i
// -> one contiguous 32-byte load per lane.
__device__ __forceinline__ v16h load_b_frag(const _Float16* __restrict__ Bt,
                                            int col0, int k0, int lane) {
  const int nn = col0 + (lane & 15);
  const int kk = k0 + ((lane >> 4) << 4);
  return *(const v16h*)(Bt + (size_t)nn * N + kk);
}

__device__ __forceinline__ v8f zero8() {
  v8f z = {0.f, 0.f, 0.f, 0.f, 0.f, 0.f, 0.f, 0.f};
  return z;
}

// ---------------------------------------------------------------- GEMM 1
// T[h] = G1 @ S_h ; S stored channel-transposed: St[h][n][k] = S_h[k][n]
__global__ __launch_bounds__(256) void gemm1_kernel(
    const _Float16* __restrict__ G1h, const _Float16* __restrict__ St,
    _Float16* __restrict__ T) {
  const int lane = threadIdx.x & 31;
  const int wave = threadIdx.x >> 5;
  const int tile = blockIdx.x * WPB + wave;
  const int ti = tile / TILES, tj = tile % TILES;
  const int h = blockIdx.y;
  const _Float16* B = St + (size_t)h * NN;
  _Float16* Tc = T + (size_t)h * NN;

  v8f acc = zero8();
  for (int k0 = 0; k0 < N; k0 += 32) {
    if (k0 + 32 < N) {
      __builtin_prefetch(G1h + (size_t)(ti * 16 + (lane & 15)) * N + k0 + 32, 0, 1);
      __builtin_prefetch(B + (size_t)(tj * 16 + (lane & 15)) * N + k0 + 32, 0, 1);
    }
    v16h a = load_a_frag(G1h, ti * 16, k0, lane);
    v16h b = load_b_frag(B, tj * 16, k0, lane);
    acc = wmma_f32_16x16x32(a, b, acc);
  }
  // C/D layout: VGPR g -> M = g + 8*(lane>>4), N = lane&15
  const int m0 = ti * 16 + ((lane >> 4) << 3);
  const int nn = tj * 16 + (lane & 15);
#pragma unroll
  for (int g = 0; g < 8; ++g)
    Tc[(size_t)(m0 + g) * N + nn] = (_Float16)acc[g];
}

// ---------------------------------------------------------------- GEMM 2 fused
// H_h = T[h] @ G2^T  (B[k][n] = G2[n][k] -> row-major G2 gives contiguous B)
// Fused epilogue: match = sum_h H_h*Wr[h] + br   -> match (f32 NxN)
//                 X_h'  = (sum_h H_h*Wm[h][h'] + bm[h'])/1000
//                 stored transposed f16: snext[h'][n][m] (contiguous v8h/lane)
__global__ __launch_bounds__(256) void gemm2_fused_kernel(
    const _Float16* __restrict__ T, const _Float16* __restrict__ G2h, int hid_in,
    const float* __restrict__ wr, const float* __restrict__ br,
    const float* __restrict__ wm, const float* __restrict__ bm,
    float* __restrict__ match, _Float16* __restrict__ snext, int do_x) {
  const int lane = threadIdx.x & 31;
  const int wave = threadIdx.x >> 5;
  const int tile = blockIdx.x * WPB + wave;
  const int ti = tile / TILES, tj = tile % TILES;

  v8f acc[HID];
  for (int h = 0; h < HID; ++h) acc[h] = zero8();

  for (int k0 = 0; k0 < N; k0 += 32) {
    if (k0 + 32 < N)
      __builtin_prefetch(G2h + (size_t)(tj * 16 + (lane & 15)) * N + k0 + 32, 0, 1);
    v16h b = load_b_frag(G2h, tj * 16, k0, lane);  // shared across channels
    for (int h = 0; h < hid_in; ++h) {
      v16h a = load_a_frag(T + (size_t)h * NN, ti * 16, k0, lane);
      acc[h] = wmma_f32_16x16x32(a, b, acc[h]);
    }
  }

  const int m0 = ti * 16 + ((lane >> 4) << 3);
  const int nn = tj * 16 + (lane & 15);

  float wrl[HID];
  for (int h = 0; h < hid_in; ++h) wrl[h] = wr[h];
  const float brv = br[0];
#pragma unroll
  for (int g = 0; g < 8; ++g) {
    float s = brv;
    for (int h = 0; h < hid_in; ++h) s += acc[h][g] * wrl[h];
    match[(size_t)(m0 + g) * N + nn] = s;
  }
  if (do_x) {
    for (int hp = 0; hp < HID - 1; ++hp) {
      const float bmv = bm[hp];
      v8h xs;
#pragma unroll
      for (int g = 0; g < 8; ++g) {
        float x = bmv;
        for (int h = 0; h < hid_in; ++h) x += acc[h][g] * wm[h * (HID - 1) + hp];
        xs[g] = (_Float16)(x * 1e-3f);
      }
      *(v8h*)(snext + (size_t)hp * NN + (size_t)nn * N + m0) = xs;  // transposed
    }
  }
}

// ---------------------------------------------------------------- small kernels

__global__ void f32_to_f16_kernel(const float* __restrict__ src,
                                  _Float16* __restrict__ dst, long long n) {
  long long i = (long long)blockIdx.x * blockDim.x + threadIdx.x;
  if (i < n) dst[i] = (_Float16)src[i];
}

__global__ void fill_u32_kernel(unsigned* __restrict__ p, unsigned val, long long n) {
  long long i = (long long)blockIdx.x * blockDim.x + threadIdx.x;
  if (i < n) p[i] = val;
}

__global__ void init_seed_kernel(const int* __restrict__ seeds,
                                 _Float16* __restrict__ St0,
                                 int* __restrict__ rowSeed, int* __restrict__ colSeed) {
  int k = blockIdx.x * blockDim.x + threadIdx.x;
  if (k < NSEEDS) {
    int r = seeds[k], c = seeds[NSEEDS + k];
    rowSeed[r] = 1;
    colSeed[c] = 1;
    St0[(size_t)c * N + r] = (_Float16)1.0f;  // transposed S_0
  }
}

__global__ __launch_bounds__(256) void row_stats_kernel(
    const float* __restrict__ M, float* __restrict__ rmax, float* __restrict__ rsum) {
  const int i = blockIdx.x, tid = threadIdx.x;
  __shared__ float s[256];
  float mx = -__builtin_inff();
  for (int j = tid; j < N; j += 256) mx = fmaxf(mx, M[(size_t)i * N + j]);
  s[tid] = mx; __syncthreads();
  for (int k = 128; k > 0; k >>= 1) { if (tid < k) s[tid] = fmaxf(s[tid], s[tid + k]); __syncthreads(); }
  const float rm = s[0]; __syncthreads();
  float sm = 0.f;
  for (int j = tid; j < N; j += 256) sm += expf(M[(size_t)i * N + j] - rm);
  s[tid] = sm; __syncthreads();
  for (int k = 128; k > 0; k >>= 1) { if (tid < k) s[tid] += s[tid + k]; __syncthreads(); }
  if (tid == 0) { rmax[i] = rm; rsum[i] = s[0]; }
}

__global__ __launch_bounds__(256) void col_stats_kernel(
    const float* __restrict__ M, float* __restrict__ cmax, float* __restrict__ csum) {
  const int j = blockIdx.x, tid = threadIdx.x;
  __shared__ float s[256];
  float mx = -__builtin_inff();
  for (int i = tid; i < N; i += 256) mx = fmaxf(mx, M[(size_t)i * N + j]);
  s[tid] = mx; __syncthreads();
  for (int k = 128; k > 0; k >>= 1) { if (tid < k) s[tid] = fmaxf(s[tid], s[tid + k]); __syncthreads(); }
  const float cm = s[0]; __syncthreads();
  float sm = 0.f;
  for (int i = tid; i < N; i += 256) sm += expf(M[(size_t)i * N + j] - cm);
  s[tid] = sm; __syncthreads();
  for (int k = 128; k > 0; k >>= 1) { if (tid < k) s[tid] += s[tid + k]; __syncthreads(); }
  if (tid == 0) { cmax[j] = cm; csum[j] = s[0]; }
}

__global__ void mn_kernel(const float* __restrict__ M,
                          const float* __restrict__ rmax, const float* __restrict__ rsum,
                          const float* __restrict__ cmax, const float* __restrict__ csum,
                          const int* __restrict__ rowSeed, const int* __restrict__ colSeed,
                          float* __restrict__ Y) {
  long long idx = (long long)blockIdx.x * blockDim.x + threadIdx.x;
  if (idx >= (long long)NN) return;
  int i = (int)(idx / N), j = (int)(idx % N);
  float m = M[idx];
  float a = expf(m - rmax[i]) / rsum[i];
  float b = expf(m - cmax[j]) / csum[j];
  float v = (a + b) * 5.0f;  // *0.5 * 10
  if (rowSeed[i] | colSeed[j]) v = 0.f;
  Y[idx] = v;
}

__global__ void seed10_kernel(const int* __restrict__ seeds, float* __restrict__ Y) {
  int k = blockIdx.x * blockDim.x + threadIdx.x;
  if (k < NSEEDS) Y[(size_t)seeds[k] * N + seeds[NSEEDS + k]] = 10.0f;
}

__global__ void newseeds_kernel(const int* __restrict__ col, const float* __restrict__ Y,
                                _Float16* __restrict__ S7) {
  int r = blockIdx.x * blockDim.x + threadIdx.x;
  if (r < N) {
    int c = col[r];
    S7[(size_t)c * N + r] = (_Float16)Y[(size_t)r * N + c];  // transposed store
  }
}

// ---------------------------------------------------------------- Hungarian LSA
// Shortest augmenting path (e-maxx), cost = -Mn (maximize). Single workgroup;
// inner column sweep + argmin parallelized across 256 threads.
__global__ __launch_bounds__(256) void hungarian_kernel(
    const float* __restrict__ Mn, int* __restrict__ colOut,
    double* __restrict__ u, double* __restrict__ v, double* __restrict__ minv,
    int* __restrict__ p, int* __restrict__ way, int* __restrict__ used) {
  const int n = N, m = N;
  const int tid = threadIdx.x, nt = blockDim.x;
  __shared__ double sval[256];
  __shared__ int sidx[256];
  __shared__ int sj0, si0, sj1, sdone;
  __shared__ double sdelta;

  for (int j = tid; j <= m; j += nt) { u[j] = 0.0; v[j] = 0.0; p[j] = 0; way[j] = 0; }
  __syncthreads();

  for (int i = 1; i <= n; ++i) {
    if (tid == 0) { p[0] = i; sj0 = 0; }
    for (int j = tid; j <= m; j += nt) { minv[j] = __builtin_inf(); used[j] = 0; }
    __syncthreads();
    while (true) {
      if (tid == 0) { used[sj0] = 1; si0 = p[sj0]; }
      __syncthreads();
      const int j0 = sj0;
      const int i0 = si0;
      const double ui0 = u[i0];
      const float* crow = Mn + (size_t)(i0 - 1) * N;
      double best = __builtin_inf(); int bestj = m + 1;
      for (int j = 1 + tid; j <= m; j += nt) {
        if (!used[j]) {
          double cur = -(double)crow[j - 1] - ui0 - v[j];
          double mv = minv[j];
          if (cur < mv) { mv = cur; minv[j] = cur; way[j] = j0; }
          if (mv < best) { best = mv; bestj = j; }  // ascending j -> first-min kept
        }
      }
      sval[tid] = best; sidx[tid] = bestj;
      __syncthreads();
      for (int s = nt >> 1; s > 0; s >>= 1) {
        if (tid < s) {
          double ov = sval[tid + s]; int oj = sidx[tid + s];
          if (ov < sval[tid] || (ov == sval[tid] && oj < sidx[tid])) {
            sval[tid] = ov; sidx[tid] = oj;
          }
        }
        __syncthreads();
      }
      if (tid == 0) { sdelta = sval[0]; sj1 = sidx[0]; }
      __syncthreads();
      const double delta = sdelta;
      const int j1 = sj1;
      for (int j = tid; j <= m; j += nt) {
        if (used[j]) { u[p[j]] += delta; v[j] -= delta; }  // distinct p[j] for used j
        else minv[j] -= delta;
      }
      __syncthreads();
      if (tid == 0) { sj0 = j1; sdone = (p[j1] == 0) ? 1 : 0; }
      __syncthreads();
      if (sdone) break;
    }
    if (tid == 0) {  // augment along stored path
      int j0 = sj0;
      while (j0 != 0) { int j1x = way[j0]; p[j0] = p[j1x]; j0 = j1x; }
    }
    __syncthreads();
  }
  for (int j = 1 + tid; j <= m; j += nt) colOut[p[j] - 1] = j - 1;
}

// ---------------------------------------------------------------- host driver

extern "C" void kernel_launch(void* const* d_in, const int* in_sizes, int n_in,
                              void* d_out, int out_size, void* d_ws, size_t ws_size,
                              hipStream_t stream) {
  (void)in_sizes; (void)n_in; (void)out_size; (void)ws_size;
  const float* G1     = (const float*)d_in[0];
  const float* G2     = (const float*)d_in[1];
  const int*   seeds  = (const int*)d_in[2];
  const float* mlp0_w = (const float*)d_in[3];   // (1,7)
  const float* mlp0_b = (const float*)d_in[4];   // (7)
  const float* mlp_w  = (const float*)d_in[5];   // (3,8,7)
  const float* mlp_b  = (const float*)d_in[6];   // (3,7)
  const float* ro0_w  = (const float*)d_in[7];   // (1,1)
  const float* ro0_b  = (const float*)d_in[8];   // (1)
  const float* ro_w   = (const float*)d_in[9];   // (3,8,1)
  const float* ro_b   = (const float*)d_in[10];  // (3,1)

  // workspace carve-up (256B aligned)
  char* base = (char*)d_ws;
  size_t off = 0;
  auto give = [&](size_t bytes) -> void* {
    off = (off + 255) & ~(size_t)255;
    void* p = base + off;
    off += bytes;
    return p;
  };
  _Float16* G1h   = (_Float16*)give(NN * 2);
  _Float16* G2h   = (_Float16*)give(NN * 2);
  _Float16* T     = (_Float16*)give((size_t)HID * NN * 2);
  _Float16* St    = (_Float16*)give((size_t)HID * NN * 2);  // channel-transposed S
  float*    match = (float*)give(NN * 4);
  float*    rmax  = (float*)give(N * 4);
  float*    rsum  = (float*)give(N * 4);
  float*    cmax  = (float*)give(N * 4);
  float*    csum  = (float*)give(N * 4);
  int*      rowSeed = (int*)give(N * 4);
  int*      colSeed = (int*)give(N * 4);
  int*      colIdx  = (int*)give(N * 4);
  double*   hu_u    = (double*)give((N + 1) * 8);
  double*   hu_v    = (double*)give((N + 1) * 8);
  double*   hu_minv = (double*)give((N + 1) * 8);
  int*      hu_p    = (int*)give((N + 1) * 4);
  int*      hu_way  = (int*)give((N + 1) * 4);
  int*      hu_used = (int*)give((N + 1) * 4);

  const long long nnll = (long long)NN;
  const int elemBlocks = (int)((nnll + 255) / 256);

  f32_to_f16_kernel<<<elemBlocks, 256, 0, stream>>>(G1, G1h, nnll);
  f32_to_f16_kernel<<<elemBlocks, 256, 0, stream>>>(G2, G2h, nnll);
  fill_u32_kernel<<<(N + 255) / 256, 256, 0, stream>>>((unsigned*)rowSeed, 0u, N);
  fill_u32_kernel<<<(N + 255) / 256, 256, 0, stream>>>((unsigned*)colSeed, 0u, N);
  fill_u32_kernel<<<(int)((nnll / 2 + 255) / 256), 256, 0, stream>>>(
      (unsigned*)St, 0u, nnll / 2);  // zero channel 0 (f16 zeros)
  init_seed_kernel<<<1, 256, 0, stream>>>(seeds, St, rowSeed, colSeed);

  for (int layer = 0; layer < 4; ++layer) {
    const int hid_in = (layer == 0) ? 1 : HID;
    const float *wr, *brp, *wm, *bmp;
    if (layer == 0) { wr = ro0_w; brp = ro0_b; wm = mlp0_w; bmp = mlp0_b; }
    else {
      wr  = ro_w + (size_t)(layer - 1) * HID;
      brp = ro_b + (layer - 1);
      wm  = mlp_w + (size_t)(layer - 1) * HID * (HID - 1);
      bmp = mlp_b + (size_t)(layer - 1) * (HID - 1);
    }
    const int do_x = (layer < 3) ? 1 : 0;

    gemm1_kernel<<<dim3(TILE2 / WPB, hid_in), 256, 0, stream>>>(G1h, St, T);
    gemm2_fused_kernel<<<TILE2 / WPB, 256, 0, stream>>>(T, G2h, hid_in, wr, brp,
                                                        wm, bmp, match, St, do_x);
    row_stats_kernel<<<N, 256, 0, stream>>>(match, rmax, rsum);
    col_stats_kernel<<<N, 256, 0, stream>>>(match, cmax, csum);

    float* Y = (float*)d_out + (size_t)layer * NN;
    mn_kernel<<<elemBlocks, 256, 0, stream>>>(match, rmax, rsum, cmax, csum,
                                              rowSeed, colSeed, Y);
    seed10_kernel<<<1, 256, 0, stream>>>(seeds, Y);

    if (do_x) {
      hungarian_kernel<<<1, 256, 0, stream>>>(Y, colIdx, hu_u, hu_v, hu_minv,
                                              hu_p, hu_way, hu_used);
      fill_u32_kernel<<<(int)((nnll / 2 + 255) / 256), 256, 0, stream>>>(
          (unsigned*)(St + (size_t)(HID - 1) * NN), 0u, nnll / 2);
      newseeds_kernel<<<(N + 255) / 256, 256, 0, stream>>>(
          colIdx, Y, St + (size_t)(HID - 1) * NN);
    }
  }
}
